// GT_62612033241315
// MI455X (gfx1250) — compile-verified, matrix operand order checked by reading, not run on hardware
//
#include <hip/hip_runtime.h>
#include <hip/hip_bf16.h>

#define N_NODES 50000
#define N_EDGES 800000
#define IN_DIM  128
#define HID     64
#define NHEAD   2
#define NLAYER  2
#define OUT_DIM 40

typedef __attribute__((ext_vector_type(16))) __bf16 v16bf;
typedef __attribute__((ext_vector_type(8)))  float  v8f;

// ---------------------------------------------------------------------------
// WMMA GEMM, row-strip version: one wave32 computes a 16-row strip across ALL
// n-tiles (Nc/16 of them). A fragment is loaded once per k-step and reused for
// every n-tile; B is staged once per BLOCK into LDS, already packed into the
// ISA fragment layout (16-bit A/B 16x32, ISA 7.12.2):
//   lane = grp*16 + mr ; element j(0..15) -> k = (j&7) + (j>=8?16:0) + (grp?8:0)
// MODE: 0 = +bias, 1 = +bias,ReLU, 2 = +bias,+resid, 3 = raw
// ---------------------------------------------------------------------------
template<int Kd, int Nc, int MODE>
__global__ void gemm_wmma_bf16(const float* __restrict__ A,
                               const float* __restrict__ B,
                               const float* __restrict__ bias,
                               const float* __restrict__ resid,
                               float* __restrict__ D, int M)
{
    constexpr int NT = Nc / 16;           // n-tiles per row strip
    constexpr int KT = Kd / 32;           // k-steps
    __shared__ __align__(32) __bf16 Bsh[KT * NT * 32 * 16];

    // ---- cooperative pack of B[k,n] (fp32, row-major) into fragment layout
    const int tid = threadIdx.x;
    for (int i = tid; i < KT * NT * 32 * 16; i += 256) {
        int j  = i & 15;
        int ln = (i >> 4) & 31;
        int nt = (i >> 9) % NT;
        int kt = (i >> 9) / NT;
        int k  = kt * 32 + (j & 7) + ((j >= 8) ? 16 : 0) + ((ln >> 4) ? 8 : 0);
        int n  = nt * 16 + (ln & 15);
        Bsh[i] = (__bf16)B[(size_t)k * Nc + n];
    }
    __syncthreads();

    const int wave = (int)((blockIdx.x * (unsigned)blockDim.x + tid) >> 5);
    const int lane = tid & 31;
    const int m0   = wave << 4;
    if (m0 >= M) return;                  // wave-uniform: EXEC stays all-ones

    const int grp = lane >> 4;            // 0 | 1
    const int mr  = lane & 15;            // row within tile (A) / column (B)

    v8f acc[NT];
#pragma unroll
    for (int nt = 0; nt < NT; ++nt) acc[nt] = v8f{};

#pragma unroll
    for (int kt = 0; kt < KT; ++kt) {
        v16bf a;
        const float* arow = A + (size_t)(m0 + mr) * Kd + kt * 32 + grp * 8;
#pragma unroll
        for (int j = 0; j < 8; ++j) a[j]     = (__bf16)arow[j];
#pragma unroll
        for (int j = 0; j < 8; ++j) a[8 + j] = (__bf16)arow[16 + j];

#pragma unroll
        for (int nt = 0; nt < NT; ++nt) {
            const v16bf b = *(const v16bf*)&Bsh[((kt * NT + nt) * 32 + lane) * 16];
            acc[nt] = __builtin_amdgcn_wmma_f32_16x16x32_bf16(
                          false, a, false, b, (short)0, acc[nt], false, false);
        }
    }

    // C/D layout: lane holds column n; VGPR r -> row m0 + r + grp*8
#pragma unroll
    for (int nt = 0; nt < NT; ++nt) {
        const int nc = nt * 16 + mr;
        const float bv = (MODE == 3) ? 0.f : bias[nc];
#pragma unroll
        for (int r = 0; r < 8; ++r) {
            const int mm = m0 + r + grp * 8;
            float v = acc[nt][r] + bv;
            if (MODE == 1) v = fmaxf(v, 0.f);
            if (MODE == 2) v += resid[(size_t)mm * Nc + nc];
            D[(size_t)mm * Nc + nc] = v;
        }
    }
}

// ---------------------------------------------------------------------------
// Edge attention logits: alpha[e,h] = dot(Q[dst,h,:], K[src,h,:]) / 8
// wave-per-edge: coalesced 128-float row reads, shuffle reduction.
// ---------------------------------------------------------------------------
__global__ void edge_alpha_k(const float* __restrict__ Q,
                             const float* __restrict__ Km,
                             const int* __restrict__ src,
                             const int* __restrict__ dst,
                             float* __restrict__ alpha, int E)
{
    int wave = (int)((blockIdx.x * (unsigned)blockDim.x + threadIdx.x) >> 5);
    int lane = threadIdx.x & 31;
    if (wave >= E) return;
    int s = src[wave], d = dst[wave];
    const float* q = Q  + (size_t)d * (NHEAD * HID);
    const float* k = Km + (size_t)s * (NHEAD * HID);
    float a0 = q[lane] * k[lane] + q[lane + 32] * k[lane + 32];
    float a1 = q[lane + 64] * k[lane + 64] + q[lane + 96] * k[lane + 96];
#pragma unroll
    for (int m = 16; m; m >>= 1) {
        a0 += __shfl_xor(a0, m, 32);
        a1 += __shfl_xor(a1, m, 32);
    }
    if (lane == 0) {
        alpha[(size_t)wave * 2]     = a0 * 0.125f;
        alpha[(size_t)wave * 2 + 1] = a1 * 0.125f;
    }
}

// monotonic float<->uint mapping so unsigned atomicMax == float max
__device__ __forceinline__ unsigned f2key(float f) {
    unsigned u = __float_as_uint(f);
    return (u & 0x80000000u) ? ~u : (u | 0x80000000u);
}
__device__ __forceinline__ float key2f(unsigned k) {
    unsigned u = (k & 0x80000000u) ? (k & 0x7FFFFFFFu) : ~k;
    return __uint_as_float(u);
}

__global__ void seg_max_k(const float* __restrict__ alpha,
                          const int* __restrict__ dst,
                          unsigned* __restrict__ amaxk, int E)
{
    int t = blockIdx.x * blockDim.x + threadIdx.x;
    if (t >= 2 * E) return;
    int e = t >> 1, h = t & 1;
    atomicMax(&amaxk[(size_t)dst[e] * 2 + h], f2key(alpha[t]));
}

__global__ void edge_exp_k(float* __restrict__ alpha,      // in: logits, out: exp
                           const int* __restrict__ dst,
                           const unsigned* __restrict__ amaxk,
                           float* __restrict__ denom, int E)
{
    int t = blockIdx.x * blockDim.x + threadIdx.x;
    if (t >= 2 * E) return;
    int e = t >> 1, h = t & 1;
    int d = dst[e];
    float v = expf(alpha[t] - key2f(amaxk[(size_t)d * 2 + h]));
    alpha[t] = v;
    atomicAdd(&denom[(size_t)d * 2 + h], v);
}

// aggr[dst,h,:] += V[src,h,:] * ea[e,h]/(denom[dst,h]+eps); wave-per-edge.
__global__ void edge_msg_k(const float* __restrict__ ea,
                           const float* __restrict__ V,
                           const int* __restrict__ src,
                           const int* __restrict__ dst,
                           const float* __restrict__ denom,
                           float* __restrict__ aggr, int E)
{
    int wave = (int)((blockIdx.x * (unsigned)blockDim.x + threadIdx.x) >> 5);
    int lane = threadIdx.x & 31;
    if (wave >= E) return;
    int s = src[wave], d = dst[wave];
    float w0 = ea[(size_t)wave * 2]     / (denom[(size_t)d * 2]     + 1e-16f);
    float w1 = ea[(size_t)wave * 2 + 1] / (denom[(size_t)d * 2 + 1] + 1e-16f);
    const float* vrow = V    + (size_t)s * (NHEAD * HID);
    float*       arow = aggr + (size_t)d * (NHEAD * HID);
#pragma unroll
    for (int it = 0; it < 4; ++it) {
        int idx = lane + 32 * it;
        atomicAdd(&arow[idx], vrow[idx] * (idx < HID ? w0 : w1));
    }
}

// h = LayerNorm(t + h) * g + b   (wave-per-row, 64 elems = 2/lane)
__global__ void ln_k(const float* __restrict__ t, float* __restrict__ h,
                     const float* __restrict__ g, const float* __restrict__ b, int M)
{
    int wave = (int)((blockIdx.x * (unsigned)blockDim.x + threadIdx.x) >> 5);
    int lane = threadIdx.x & 31;
    if (wave >= M) return;
    const float* tr = t + (size_t)wave * HID;
    float* hr = h + (size_t)wave * HID;
    float x0 = tr[lane] + hr[lane];
    float x1 = tr[lane + 32] + hr[lane + 32];
    float s = x0 + x1;
#pragma unroll
    for (int m = 16; m; m >>= 1) s += __shfl_xor(s, m, 32);
    float mu = s * (1.f / HID);
    float d0 = x0 - mu, d1 = x1 - mu;
    float v = d0 * d0 + d1 * d1;
#pragma unroll
    for (int m = 16; m; m >>= 1) v += __shfl_xor(v, m, 32);
    float rstd = rsqrtf(v * (1.f / HID) + 1e-5f);
    hr[lane]      = d0 * rstd * g[lane]      + b[lane];
    hr[lane + 32] = d1 * rstd * g[lane + 32] + b[lane + 32];
}

// out = log_softmax(h @ W[64,40] + b); wave-per-row (lane owns cols lane, lane+32)
__global__ void head_lsm_k(const float* __restrict__ h,
                           const float* __restrict__ W,
                           const float* __restrict__ b,
                           float* __restrict__ out, int M)
{
    int wave = (int)((blockIdx.x * (unsigned)blockDim.x + threadIdx.x) >> 5);
    int lane = threadIdx.x & 31;
    if (wave >= M) return;
    const float* hr = h + (size_t)wave * HID;
    int c1 = lane + 32;
    float a0 = 0.f, a1 = 0.f;
#pragma unroll 8
    for (int k = 0; k < HID; ++k) {
        float hv = hr[k];
        a0 += hv * W[k * OUT_DIM + lane];
        if (c1 < OUT_DIM) a1 += hv * W[k * OUT_DIM + c1];
    }
    a0 += b[lane];
    if (c1 < OUT_DIM) a1 += b[c1];
    float mx = (c1 < OUT_DIM) ? fmaxf(a0, a1) : a0;
#pragma unroll
    for (int m = 16; m; m >>= 1) mx = fmaxf(mx, __shfl_xor(mx, m, 32));
    float se = expf(a0 - mx) + ((c1 < OUT_DIM) ? expf(a1 - mx) : 0.f);
#pragma unroll
    for (int m = 16; m; m >>= 1) se += __shfl_xor(se, m, 32);
    float lse = logf(se);
    out[(size_t)wave * OUT_DIM + lane] = a0 - mx - lse;
    if (c1 < OUT_DIM) out[(size_t)wave * OUT_DIM + c1] = a1 - mx - lse;
}

// ---------------------------------------------------------------------------
extern "C" void kernel_launch(void* const* d_in, const int* in_sizes, int n_in,
                              void* d_out, int out_size, void* d_ws, size_t ws_size,
                              hipStream_t stream)
{
    const int N = N_NODES, E = N_EDGES;
    const float* x     = (const float*)d_in[0];
    const int*   ei    = (const int*)d_in[1];
    const float* emb_w = (const float*)d_in[2];
    const float* emb_b = (const float*)d_in[3];
    const float* Wq    = (const float*)d_in[4];
    const float* Wk    = (const float*)d_in[5];
    const float* Wv    = (const float*)d_in[6];
    const float* Wout  = (const float*)d_in[7];
    const float* bout  = (const float*)d_in[8];
    const float* ln_g  = (const float*)d_in[9];
    const float* ln_b  = (const float*)d_in[10];
    const float* W1    = (const float*)d_in[11];
    const float* b1    = (const float*)d_in[12];
    const float* W2    = (const float*)d_in[13];
    const float* b2    = (const float*)d_in[14];
    const float* out_w = (const float*)d_in[15];
    const float* out_b = (const float*)d_in[16];

    const int* src = ei;
    const int* dst = ei + E;

    // workspace layout (floats)
    float* ws = (float*)d_ws;
    size_t off = 0;
    float*    h     = ws + off; off += (size_t)N * HID;
    float*    t0    = ws + off; off += (size_t)N * HID;
    float*    Qb    = ws + off; off += (size_t)N * NHEAD * HID;
    float*    Kb    = ws + off; off += (size_t)N * NHEAD * HID;
    float*    Vb    = ws + off; off += (size_t)N * NHEAD * HID;
    float*    aggr  = ws + off; off += (size_t)N * NHEAD * HID;
    float*    alpha = ws + off; off += (size_t)E * NHEAD;
    float*    denom = ws + off; off += (size_t)N * NHEAD;
    unsigned* amaxk = (unsigned*)(ws + off); off += (size_t)N * NHEAD;
    if (off * sizeof(float) > ws_size) return;   // workspace too small: bail

    const int gemmBlocks  = ((N / 16) + 7) / 8;    // 8 waves (row strips) / block
    const int waveBlocksN = (N + 7) / 8;           // wave-per-row kernels
    const int waveBlocksE = (E + 7) / 8;           // wave-per-edge kernels
    const int thrBlocks2E = (2 * E + 255) / 256;   // thread-per-(edge,head)

    // h = x @ emb_w + emb_b
    gemm_wmma_bf16<IN_DIM, HID, 0><<<gemmBlocks, 256, 0, stream>>>(
        x, emb_w, emb_b, nullptr, h, N);

    for (int l = 0; l < NLAYER; ++l) {
        const float* Wq_l = Wq + (size_t)l * HID * NHEAD * HID;
        const float* Wk_l = Wk + (size_t)l * HID * NHEAD * HID;
        const float* Wv_l = Wv + (size_t)l * HID * NHEAD * HID;
        const float* Wo_l = Wout + (size_t)l * NHEAD * HID * HID;

        gemm_wmma_bf16<HID, NHEAD * HID, 3><<<gemmBlocks, 256, 0, stream>>>(
            h, Wq_l, nullptr, nullptr, Qb, N);
        gemm_wmma_bf16<HID, NHEAD * HID, 3><<<gemmBlocks, 256, 0, stream>>>(
            h, Wk_l, nullptr, nullptr, Kb, N);
        gemm_wmma_bf16<HID, NHEAD * HID, 3><<<gemmBlocks, 256, 0, stream>>>(
            h, Wv_l, nullptr, nullptr, Vb, N);

        hipMemsetAsync(amaxk, 0, (size_t)N * NHEAD * sizeof(unsigned), stream);
        hipMemsetAsync(denom, 0, (size_t)N * NHEAD * sizeof(float), stream);
        hipMemsetAsync(aggr,  0, (size_t)N * NHEAD * HID * sizeof(float), stream);

        edge_alpha_k<<<waveBlocksE, 256, 0, stream>>>(Qb, Kb, src, dst, alpha, E);
        seg_max_k  <<<thrBlocks2E, 256, 0, stream>>>(alpha, dst, amaxk, E);
        edge_exp_k <<<thrBlocks2E, 256, 0, stream>>>(alpha, dst, amaxk, denom, E);
        edge_msg_k <<<waveBlocksE, 256, 0, stream>>>(alpha, Vb, src, dst, denom, aggr, E);

        // t0 = aggr @ Wout + bout ; h = LN(t0 + h)
        gemm_wmma_bf16<NHEAD * HID, HID, 0><<<gemmBlocks, 256, 0, stream>>>(
            aggr, Wo_l, bout + l * HID, nullptr, t0, N);
        ln_k<<<waveBlocksN, 256, 0, stream>>>(t0, h, ln_g + l * HID, ln_b + l * HID, N);

        // FFN: t0 = relu(h@W1+b1); h = h + t0@W2 + b2
        gemm_wmma_bf16<HID, HID, 1><<<gemmBlocks, 256, 0, stream>>>(
            h,  W1 + (size_t)l * HID * HID, b1 + l * HID, nullptr, t0, N);
        gemm_wmma_bf16<HID, HID, 2><<<gemmBlocks, 256, 0, stream>>>(
            t0, W2 + (size_t)l * HID * HID, b2 + l * HID, h, h, N);
    }

    head_lsm_k<<<waveBlocksN, 256, 0, stream>>>(h, out_w, out_b, (float*)d_out, N);
}
